// MicroStepDecoder_50105088475266
// MI455X (gfx1250) — matrix operand order repaired and checked by reading, not compiled, not run on hardware
//
#include <hip/hip_runtime.h>
#include <hip/hip_bf16.h>
#include <math.h>

// ---------------- types ----------------
typedef __bf16 bf16;
typedef __attribute__((ext_vector_type(16))) __bf16 v16bf;
typedef __attribute__((ext_vector_type(8)))  __bf16 v8bf;
typedef __attribute__((ext_vector_type(8)))  float  v8f;

#define HEADS    32
#define KV_HEADS 8
#define HEAD_DIM 64
#define DMODEL   2048
#define DFF      8192
#define NTOK     2048   // B*(S-I) = 8*256
#define TMAX     6
#define QKV_N    3072   // 2048 q + 512 k + 512 v

// ---------------- CDNA5 async copy helpers ----------------
// 16B per lane, global -> LDS, tracked by ASYNCcnt (no VGPR staging).
#define ASYNC_CP16(ldsoff, gp)                                              \
  asm volatile("global_load_async_to_lds_b128 %0, %1, off"                  \
               :: "v"((unsigned)(ldsoff)),                                  \
                  "v"((unsigned long long)(uintptr_t)(gp))                  \
               : "memory")

template <int N>
__device__ __forceinline__ void wait_async() {
#if defined(__has_builtin) && __has_builtin(__builtin_amdgcn_s_wait_asynccnt)
  __builtin_amdgcn_s_wait_asynccnt((short)N);
#else
  asm volatile("s_wait_asynccnt %0" :: "i"(N) : "memory");
#endif
}

// ---------------- weight transpose + bf16 convert: src[K][N] f32 -> dst[N][K] bf16 ----------------
__global__ void k_transpose_bf16(const float* __restrict__ src, bf16* __restrict__ dst,
                                 int K, int N) {
  __shared__ float tile[32][33];
  int n0 = blockIdx.x * 32, k0 = blockIdx.y * 32;
  int tx = threadIdx.x, ty = threadIdx.y;              // 32 x 8
  #pragma unroll
  for (int i = 0; i < 32; i += 8)
    tile[ty + i][tx] = src[(size_t)(k0 + ty + i) * N + n0 + tx];
  __syncthreads();
  #pragma unroll
  for (int i = 0; i < 32; i += 8)
    dst[(size_t)(n0 + ty + i) * K + k0 + tx] = (bf16)tile[tx][ty + i];
}

// ---------------- gather instruction tail tokens into padded hidden buffer ----------------
__global__ void k_gather(const float* __restrict__ hs, const long long* __restrict__ seq,
                         const long long* __restrict__ inst, int B, float* __restrict__ hpad) {
  int n = blockIdx.x;
  long long off = 0, s = 0, ii = 0;
  int nn = n;
  for (int b = 0; b < B; ++b) {
    s = seq[b]; ii = inst[b];
    int cnt = (int)(s - ii);
    if (nn < cnt) break;
    nn -= cnt; off += s;
  }
  long long src = off + ii - 1 + nn;
  const float* sp = hs + (size_t)src * DMODEL;
  float* dp = hpad + (size_t)n * TMAX * DMODEL;
  for (int d = threadIdx.x; d < DMODEL; d += blockDim.x) dp[d] = sp[d];
}

// ---------------- RMSNorm (padded source) -> bf16 compact ----------------
__global__ void k_rmsnorm_bf16(const float* __restrict__ hpad, const float* __restrict__ w,
                               bf16* __restrict__ dst, int t) {
  int m = blockIdx.x, tid = threadIdx.x;
  int n = m / t, j = m % t;
  const float* x = hpad + ((size_t)n * TMAX + j) * DMODEL;
  float v[8]; float ss = 0.f;
  #pragma unroll
  for (int i = 0; i < 8; ++i) { float a = x[tid + i * 256]; v[i] = a; ss += a * a; }
  __shared__ float red[256];
  red[tid] = ss; __syncthreads();
  for (int s2 = 128; s2 > 0; s2 >>= 1) { if (tid < s2) red[tid] += red[tid + s2]; __syncthreads(); }
  float scale = rsqrtf(red[0] / (float)DMODEL + 1e-6f);
  bf16* o = dst + (size_t)m * DMODEL;
  #pragma unroll
  for (int i = 0; i < 8; ++i) { int c = tid + i * 256; o[c] = (bf16)(v[i] * scale * w[c]); }
}

// ---------------- residual add + RMSNorm: hmid = hpad + oproj ; xn = rms(hmid)*w ----------------
__global__ void k_add_rmsnorm(const float* __restrict__ hpad, const float* __restrict__ oproj,
                              const float* __restrict__ w, float* __restrict__ hmid,
                              bf16* __restrict__ xn, int t) {
  int m = blockIdx.x, tid = threadIdx.x;
  int n = m / t, j = m % t;
  const float* h = hpad + ((size_t)n * TMAX + j) * DMODEL;
  const float* o = oproj + (size_t)m * DMODEL;
  float v[8]; float ss = 0.f;
  #pragma unroll
  for (int i = 0; i < 8; ++i) { int c = tid + i * 256; float a = h[c] + o[c]; v[i] = a; ss += a * a; }
  __shared__ float red[256];
  red[tid] = ss; __syncthreads();
  for (int s2 = 128; s2 > 0; s2 >>= 1) { if (tid < s2) red[tid] += red[tid + s2]; __syncthreads(); }
  float scale = rsqrtf(red[0] / (float)DMODEL + 1e-6f);
  float* hm = hmid + (size_t)m * DMODEL;
  bf16* x2 = xn + (size_t)m * DMODEL;
  #pragma unroll
  for (int i = 0; i < 8; ++i) { int c = tid + i * 256; hm[c] = v[i]; x2[c] = (bf16)(v[i] * scale * w[c]); }
}

// ---------------- RoPE in-place on QKV (q heads 0..31, k heads 32..39) ----------------
__global__ void k_rope(float* __restrict__ qkv, int t) {
  int idx = blockIdx.x * blockDim.x + threadIdx.x;
  int dp = idx & 31; int rest = idx >> 5;
  int head = rest % (HEADS + KV_HEADS);
  int m = rest / (HEADS + KV_HEADS);
  int j = m % t;
  float inv = __powf(10000.f, -(float)dp / 32.f);
  float ang = (float)j * inv;
  float c = __cosf(ang), s = __sinf(ang);
  float* base = qkv + (size_t)m * QKV_N + (head < HEADS ? head * 64 : 2048 + (head - HEADS) * 64);
  float x1 = base[dp], x2 = base[dp + 32];
  base[dp]      = x1 * c - x2 * s;
  base[dp + 32] = x2 * c + x1 * s;
}

// ---------------- tiny causal attention (t<=5), one wave32 per (n, head) ----------------
__global__ void k_attn(const float* __restrict__ qkv, bf16* __restrict__ attn_out, int t) {
  int nh = blockIdx.x;
  int n = nh >> 5, h = nh & 31, hk = h >> 2;
  int lane = threadIdx.x;                       // 0..31, dims {lane, lane+32}
  float kv[5][2], vv[5][2];
  for (int j = 0; j < t; ++j) {
    const float* row = qkv + (size_t)(n * t + j) * QKV_N;
    kv[j][0] = row[2048 + hk * 64 + lane];
    kv[j][1] = row[2048 + hk * 64 + lane + 32];
    vv[j][0] = row[2560 + hk * 64 + lane];
    vv[j][1] = row[2560 + hk * 64 + lane + 32];
  }
  for (int jq = 0; jq < t; ++jq) {
    const float* row = qkv + (size_t)(n * t + jq) * QKV_N;
    float q0 = row[h * 64 + lane], q1 = row[h * 64 + lane + 32];
    float sc[5];
    for (int j = 0; j <= jq; ++j) {
      float p = q0 * kv[j][0] + q1 * kv[j][1];
      #pragma unroll
      for (int off = 16; off > 0; off >>= 1) p += __shfl_xor(p, off, 32);
      sc[j] = p * 0.125f;                       // 1/sqrt(64)
    }
    float mx = -1e30f;
    for (int j = 0; j <= jq; ++j) mx = fmaxf(mx, sc[j]);
    float den = 0.f, o0 = 0.f, o1 = 0.f;
    for (int j = 0; j <= jq; ++j) {
      float e = __expf(sc[j] - mx);
      den += e; o0 += e * vv[j][0]; o1 += e * vv[j][1];
    }
    float r = 1.f / den;
    bf16* out = attn_out + (size_t)(n * t + jq) * DMODEL + h * 64;
    out[lane]      = (bf16)(o0 * r);
    out[lane + 32] = (bf16)(o1 * r);
  }
}

// ---------------- bf16 WMMA GEMM: C[M][N] = A[M][K] * WT[N][K]^T ----------------
// Async double-buffered LDS staging (GLOBAL_LOAD_ASYNC_TO_LDS_B128 + ASYNCcnt).
// flag 0: store f32 C.  flag 1: store bf16( silu(aux[m][n]) * acc )  (fused SiLU-up epilogue)
#define BM 128
#define BN 128
#define BKD 64
#define LDS_S 72            // padded row stride (bf16): 144B, 16B-aligned
#define BUF_ELEMS (BM * LDS_S)

__global__ __launch_bounds__(256)
void k_gemm_bf16(const bf16* __restrict__ A, const bf16* __restrict__ B,
                 void* __restrict__ C, const float* __restrict__ aux,
                 int M, int N, int K, int flag) {
  __shared__ __align__(16) bf16 As[2][BUF_ELEMS];
  __shared__ __align__(16) bf16 Bs[2][BUF_ELEMS];
  int tid = threadIdx.x;
  int lane = tid & 31, wave = tid >> 5;
  int wm = wave >> 2, wn = wave & 3;                 // 2 x 4 wave grid, 64x32 per wave
  int m0 = blockIdx.y * BM, n0 = blockIdx.x * BN;
  int half = lane >> 4, r16 = lane & 15;

  v8f acc[4][2];
  #pragma unroll
  for (int mt = 0; mt < 4; ++mt)
    #pragma unroll
    for (int nt = 0; nt < 2; ++nt) acc[mt][nt] = (v8f){};

  // per-thread staging lanes: 1024 16B-chunks per 128x64 tile -> 4 per thread per matrix
  const bf16* pa[4]; const bf16* pb[4];
  unsigned la[4], lb[4];
  #pragma unroll
  for (int it = 0; it < 4; ++it) {
    int idx = tid + it * 256;
    int r = idx >> 3, kc = idx & 7;
    pa[it] = A + (size_t)(m0 + r) * K + kc * 8;
    pb[it] = B + (size_t)(n0 + r) * K + kc * 8;
    la[it] = (unsigned)(uintptr_t)&As[0][r * LDS_S + kc * 8];   // LDS byte offset (addr[31:0])
    lb[it] = (unsigned)(uintptr_t)&Bs[0][r * LDS_S + kc * 8];
  }
  const unsigned bufbytes = BUF_ELEMS * 2;

  // prologue: stage tile 0 into buffer 0
  #pragma unroll
  for (int it = 0; it < 4; ++it) {
    ASYNC_CP16(la[it], pa[it]); ASYNC_CP16(lb[it], pb[it]);
    pa[it] += BKD; pb[it] += BKD;
  }

  int ntiles = K / BKD;
  for (int ti = 0; ti < ntiles; ++ti) {
    int cur = ti & 1;
    if (ti + 1 < ntiles) {                           // stage next tile, then wait for current
      unsigned boff = (unsigned)(cur ^ 1) * bufbytes;
      #pragma unroll
      for (int it = 0; it < 4; ++it) {
        ASYNC_CP16(la[it] + boff, pa[it]); ASYNC_CP16(lb[it] + boff, pb[it]);
        pa[it] += BKD; pb[it] += BKD;
      }
      wait_async<8>();                               // in-order: current tile's 8 ops landed
    } else {
      wait_async<0>();
    }
    __syncthreads();                                 // all waves' writes visible

    const bf16* as = As[cur];
    const bf16* bs = Bs[cur];
    #pragma unroll
    for (int ks = 0; ks < BKD; ks += 32) {
      v16bf af[4], bfr[2];
      union { v16bf v; v8bf h[2]; } u;
      #pragma unroll
      for (int mt = 0; mt < 4; ++mt) {               // A frag: K {0-7,16-23} / {8-15,24-31}
        const bf16* p = &as[(wm * 64 + mt * 16 + r16) * LDS_S + ks];
        u.h[0] = *(const v8bf*)(p + half * 8);
        u.h[1] = *(const v8bf*)(p + 16 + half * 8);
        af[mt] = u.v;
      }
      #pragma unroll
      for (int nt = 0; nt < 2; ++nt) {               // B frag: K {0-7,16-23} / {8-15,24-31}
        const bf16* p = &bs[(wn * 32 + nt * 16 + r16) * LDS_S + ks];
        u.h[0] = *(const v8bf*)(p + half * 8);
        u.h[1] = *(const v8bf*)(p + 16 + half * 8);
        bfr[nt] = u.v;
      }
      #pragma unroll
      for (int mt = 0; mt < 4; ++mt)
        #pragma unroll
        for (int nt = 0; nt < 2; ++nt)
          acc[mt][nt] = __builtin_amdgcn_wmma_f32_16x16x32_bf16(
              false, af[mt], false, bfr[nt], (short)0, acc[mt][nt], false, false);
    }
    __syncthreads();                                 // reads done before buffer is re-staged
  }

  #pragma unroll
  for (int mt = 0; mt < 4; ++mt)
    #pragma unroll
    for (int nt = 0; nt < 2; ++nt) {
      int nn = n0 + wn * 32 + nt * 16 + r16;
      #pragma unroll
      for (int g = 0; g < 8; ++g) {
        int mm = m0 + wm * 64 + mt * 16 + half * 8 + g;
        float v = acc[mt][nt][g];
        size_t o = (size_t)mm * N + nn;
        if (flag == 0) {
          ((float*)C)[o] = v;
        } else {
          float gx = aux[o];
          float sg = gx / (1.f + __expf(-gx));       // silu(gate)
          ((bf16*)C)[o] = (bf16)(sg * v);
        }
      }
    }
}

// ---------------- final residual: out = hmid + down ; append last position to hpad ----------------
__global__ void k_final_add(const float* __restrict__ hmid, const float* __restrict__ dn,
                            float* __restrict__ out, float* __restrict__ hpad, int t) {
  int m = blockIdx.x, tid = threadIdx.x;
  int n = m / t, j = m % t;
  const float* a = hmid + (size_t)m * DMODEL;
  const float* b = dn + (size_t)m * DMODEL;
  float* o = out + (size_t)m * DMODEL;
  float* ap = (j == t - 1) ? (hpad + ((size_t)n * TMAX + t) * DMODEL) : nullptr;
  #pragma unroll
  for (int i = 0; i < 8; ++i) {
    int c = tid + i * 256;
    float v = a[c] + b[c];
    o[c] = v;
    if (ap) ap[c] = v;
  }
}

// ---------------- host orchestration ----------------
extern "C" void kernel_launch(void* const* d_in, const int* in_sizes, int n_in,
                              void* d_out, int out_size, void* d_ws, size_t ws_size,
                              hipStream_t stream) {
  const float*     hs    = (const float*)d_in[0];
  const long long* seq   = (const long long*)d_in[1];
  const long long* inst  = (const long long*)d_in[2];
  const float*     w_q   = (const float*)d_in[3];
  const float*     w_k   = (const float*)d_in[4];
  const float*     w_v   = (const float*)d_in[5];
  const float*     w_o   = (const float*)d_in[6];
  const float*     ln1   = (const float*)d_in[7];
  const float*     ln2   = (const float*)d_in[8];
  const float*     w_g   = (const float*)d_in[9];
  const float*     w_u   = (const float*)d_in[10];
  const float*     w_d   = (const float*)d_in[11];
  int B = in_sizes[1];
  (void)n_in; (void)out_size; (void)ws_size;

  const size_t MMAX = (size_t)NTOK * 5;
  char* wp = (char*)d_ws;
  auto carve = [&](size_t bytes) -> void* {
    void* p = (void*)wp; wp += (bytes + 255) & ~(size_t)255; return p;
  };
  bf16*  WqkvT = (bf16*)carve((size_t)QKV_N * DMODEL * 2);
  bf16*  WoT   = (bf16*)carve((size_t)DMODEL * DMODEL * 2);
  bf16*  WgT   = (bf16*)carve((size_t)DFF * DMODEL * 2);
  bf16*  WuT   = (bf16*)carve((size_t)DFF * DMODEL * 2);
  bf16*  WdT   = (bf16*)carve((size_t)DMODEL * DFF * 2);
  float* Hpad  = (float*)carve((size_t)NTOK * TMAX * DMODEL * 4);
  bf16*  Xn    = (bf16*)carve(MMAX * DMODEL * 2);
  float* QKV   = (float*)carve(MMAX * QKV_N * 4);
  bf16*  Attn  = (bf16*)carve(MMAX * DMODEL * 2);
  float* Oproj = (float*)carve(MMAX * DMODEL * 4);   // reused for down-proj output
  float* Hmid  = (float*)carve(MMAX * DMODEL * 4);
  float* Gbuf  = (float*)carve(MMAX * DFF * 4);
  bf16*  Pbuf  = (bf16*)carve(MMAX * DFF * 2);
  float* Hout  = (float*)carve(MMAX * DMODEL * 4);

  dim3 tb(32, 8);
  // pack W_q|W_k|W_v transposed into WqkvT rows [0,2048) [2048,2560) [2560,3072)
  k_transpose_bf16<<<dim3(2048/32, 2048/32), tb, 0, stream>>>(w_q, WqkvT,                         2048, 2048);
  k_transpose_bf16<<<dim3( 512/32, 2048/32), tb, 0, stream>>>(w_k, WqkvT + (size_t)2048 * 2048,   2048,  512);
  k_transpose_bf16<<<dim3( 512/32, 2048/32), tb, 0, stream>>>(w_v, WqkvT + (size_t)2560 * 2048,   2048,  512);
  k_transpose_bf16<<<dim3(2048/32, 2048/32), tb, 0, stream>>>(w_o, WoT,  2048, 2048);
  k_transpose_bf16<<<dim3(8192/32, 2048/32), tb, 0, stream>>>(w_g, WgT,  2048, 8192);
  k_transpose_bf16<<<dim3(8192/32, 2048/32), tb, 0, stream>>>(w_u, WuT,  2048, 8192);
  k_transpose_bf16<<<dim3(2048/32, 8192/32), tb, 0, stream>>>(w_d, WdT,  8192, 2048);

  k_gather<<<NTOK, 256, 0, stream>>>(hs, seq, inst, B, Hpad);

  for (int t = 1; t <= 5; ++t) {
    int M = NTOK * t;
    k_rmsnorm_bf16<<<M, 256, 0, stream>>>(Hpad, ln1, Xn, t);

    k_gemm_bf16<<<dim3(QKV_N / BN, M / BM), 256, 0, stream>>>(Xn, WqkvT, QKV, nullptr, M, QKV_N, DMODEL, 0);
    k_rope<<<(M * (HEADS + KV_HEADS) * 32) / 256, 256, 0, stream>>>(QKV, t);
    k_attn<<<NTOK * HEADS, 32, 0, stream>>>(QKV, Attn, t);
    k_gemm_bf16<<<dim3(DMODEL / BN, M / BM), 256, 0, stream>>>(Attn, WoT, Oproj, nullptr, M, DMODEL, DMODEL, 0);

    k_add_rmsnorm<<<M, 256, 0, stream>>>(Hpad, Oproj, ln2, Hmid, Xn, t);

    k_gemm_bf16<<<dim3(DFF / BN, M / BM), 256, 0, stream>>>(Xn, WgT, Gbuf, nullptr, M, DFF, DMODEL, 0);
    k_gemm_bf16<<<dim3(DFF / BN, M / BM), 256, 0, stream>>>(Xn, WuT, Pbuf, Gbuf,   M, DFF, DMODEL, 1);
    k_gemm_bf16<<<dim3(DMODEL / BN, M / BM), 256, 0, stream>>>(Pbuf, WdT, Oproj, nullptr, M, DMODEL, DFF, 0);

    float* outp = (t == 5) ? (float*)d_out : Hout;
    k_final_add<<<M, 256, 0, stream>>>(Hmid, Oproj, outp, Hpad, t);
  }
}
